// MambaLayer_79602923864500
// MI455X (gfx1250) — matrix operand order repaired and checked by reading, not compiled, not run on hardware
//
#include <hip/hip_runtime.h>
#include <stdint.h>

#define L_TOK 16384
#define DIMC 48
#define DINNER 96
#define NCHUNK 64
#define CLEN 256

typedef __attribute__((ext_vector_type(16))) _Float16 v16h;
typedef __attribute__((ext_vector_type(8)))  float    v8f;

__device__ __forceinline__ float siluf(float x) { return x / (1.0f + __expf(-x)); }
__device__ __forceinline__ float softplusf(float x) { return (x > 20.0f) ? x : log1pf(__expf(x)); }

// ---------------------------------------------------------------------------
// Diagonal (anti-diagonal) order over the 32x32 (Y,Z) plane, matching the
// reference's stable argsort of (y+z). perm[k] = y*32+z of the k-th token.
__global__ void k_perm(int* __restrict__ perm) {
  int k = threadIdx.x;  // 1024 threads
  int T = 0, s = 0;
  for (;;) {
    int cnt = (s <= 31) ? (s + 1) : (63 - s);
    if (k < T + cnt) break;
    T += cnt; ++s;
  }
  int y = ((s > 31) ? (s - 31) : 0) + (k - T);
  int z = s - y;
  perm[k] = y * 32 + z;
}

// ---------------------------------------------------------------------------
// Weight packing to f16 (K zero-padded), dt_proj folded into x_proj, A = -exp(A_log)
__global__ void k_prep(const float* __restrict__ inw, const float* __restrict__ xpw,
                       const float* __restrict__ dtw, const float* __restrict__ outw,
                       const float* __restrict__ alog,
                       _Float16* __restrict__ W1, _Float16* __restrict__ W2,
                       _Float16* __restrict__ W3, float* __restrict__ Aneg) {
  int t = threadIdx.x;
  for (int i = t; i < 192 * 64; i += 256) {           // in_proj: 192 x 64 (pad K 48->64)
    int n = i >> 6, k = i & 63;
    W1[i] = (k < 48) ? (_Float16)inw[n * 48 + k] : (_Float16)0.0f;
  }
  for (int i = t; i < 128 * 96; i += 256) {           // fused [dt(96); B(16); C(16)] x 96
    int n = i / 96, k = i - n * 96;
    float v;
    if (n < 96)
      v = dtw[n*3+0]*xpw[0*96+k] + dtw[n*3+1]*xpw[1*96+k] + dtw[n*3+2]*xpw[2*96+k];
    else
      v = xpw[(3 + (n - 96)) * 96 + k];
    W2[i] = (_Float16)v;
  }
  for (int i = t; i < 48 * 96; i += 256) W3[i] = (_Float16)outw[i];
  for (int i = t; i < 96 * 16; i += 256) Aneg[i] = -__expf(alog[i]);
}

// ---------------------------------------------------------------------------
// Gather (diag order) + LayerNorm over 48 channels, emit f16 rows padded to 64
__global__ __launch_bounds__(256) void k_gather_ln(
    const float* __restrict__ x, const float* __restrict__ lnw,
    const float* __restrict__ lnb, const int* __restrict__ perm,
    _Float16* __restrict__ xnh) {
  int l = blockIdx.x * 256 + threadIdx.x;
  int xv = l >> 10;
  int j = perm[l & 1023];
  int y = j >> 5, z = j & 31;
  int base = (z * 32 + y) * 16 + xv;        // x layout (48, 32, 32, 16)
  float v[48];
  float mu = 0.0f;
  #pragma unroll
  for (int c = 0; c < 48; ++c) { v[c] = x[c * 16384 + base]; mu += v[c]; }
  mu *= (1.0f / 48.0f);
  float var = 0.0f;
  #pragma unroll
  for (int c = 0; c < 48; ++c) { float d = v[c] - mu; var += d * d; }
  float inv = rsqrtf(var * (1.0f / 48.0f) + 1e-5f);
  _Float16* row = xnh + (size_t)l * 64;
  #pragma unroll
  for (int c = 0; c < 48; ++c) row[c] = (_Float16)((v[c] - mu) * inv * lnw[c] + lnb[c]);
  #pragma unroll
  for (int c = 48; c < 64; ++c) row[c] = (_Float16)0.0f;
}

// ---------------------------------------------------------------------------
// Generic f16 WMMA GEMM: D[M,N] = A[M,K] * W[N,K]^T, one wave per 16x16 tile.
// Operand layouts used here (A and B share one k order, not the ISA table order):
//   A 16x32: lane m=lane&15, half=lane>>4; elem i -> K = half*16 + i (contiguous)
//   B 32x16: lane n=lane&15, half=lane>>4; elem i -> K = half*16 + i (contiguous)
//   C/D:     lane n=lane&15; VGPR r -> row m = r + 8*(lane>>4)
__global__ __launch_bounds__(256) void gemm_wmma(
    const _Float16* __restrict__ A, int lda,
    const _Float16* __restrict__ W, int ldb,
    float* __restrict__ Dst, int ldd,
    int Mtiles, int Ntiles, int Ksteps) {
  int wave = threadIdx.x >> 5;
  int lane = threadIdx.x & 31;
  int tile = blockIdx.x * 8 + wave;
  if (tile >= Mtiles * Ntiles) return;     // wave-uniform guard (EXEC all-1 for WMMA)
  int mt = tile / Ntiles, nt = tile - mt * Ntiles;
  int mn = lane & 15, half = lane >> 4;
  v8f acc = {};
  for (int ks = 0; ks < Ksteps; ++ks) {
    int k0 = ks * 32;
    const _Float16* pa = A + (size_t)(mt * 16 + mn) * lda + k0 + half * 16;
    v16h a;
    #pragma unroll
    for (int i = 0; i < 8; ++i) { a[i] = pa[i]; a[i + 8] = pa[i + 8]; }
    const _Float16* pb = W + (size_t)(nt * 16 + mn) * ldb + k0 + half * 16;
    v16h b = *(const v16h*)pb;             // 32B contiguous, aligned
    acc = __builtin_amdgcn_wmma_f32_16x16x32_f16(false, a, false, b,
                                                 (short)0, acc, false, false);
  }
  #pragma unroll
  for (int r = 0; r < 8; ++r)
    Dst[(size_t)(mt * 16 + r + half * 8) * ldd + nt * 16 + mn] = acc[r];
}

// ---------------------------------------------------------------------------
// Depthwise causal conv(4) + SiLU on xm; SiLU gate for zg
__global__ __launch_bounds__(256) void k_conv(
    const float* __restrict__ xz, const float* __restrict__ cw,
    const float* __restrict__ cb, float* __restrict__ xc,
    _Float16* __restrict__ xch, float* __restrict__ szg) {
  int idx = blockIdx.x * 256 + threadIdx.x;
  if (idx >= L_TOK * DINNER) return;
  int l = idx / DINNER, c = idx - l * DINNER;
  float acc = cb[c];
  #pragma unroll
  for (int t = 0; t < 4; ++t) {
    int li = l - 3 + t;
    if (li >= 0) acc += cw[c * 4 + t] * xz[(size_t)li * 192 + c];
  }
  float v = siluf(acc);
  xc[idx] = v;
  xch[idx] = (_Float16)v;
  szg[idx] = siluf(xz[(size_t)l * 192 + 96 + c]);
}

// softplus epilogue: dt = softplus(dbc2[:, :96] + bias); dtxc = dt*xc
__global__ __launch_bounds__(256) void k_dt(
    const float* __restrict__ dbc2, const float* __restrict__ dtb,
    const float* __restrict__ xc, float* __restrict__ dtv,
    float* __restrict__ dtxc) {
  int idx = blockIdx.x * 256 + threadIdx.x;
  if (idx >= L_TOK * DINNER) return;
  int l = idx / DINNER, c = idx - l * DINNER;
  float dt = softplusf(dbc2[(size_t)l * 128 + c] + dtb[c]);
  dtv[idx] = dt;
  dtxc[idx] = dt * xc[idx];
}

// ---------------------------------------------------------------------------
// Chunked parallel scan over h = a*h + b, 1536 independent (c,n) recurrences.
// Pass 1: per-chunk aggregates (prod a, h with zero carry-in)
__global__ __launch_bounds__(256) void k_scan1(
    const float* __restrict__ dtv, const float* __restrict__ dtxc,
    const float* __restrict__ dbc2, const float* __restrict__ Aneg,
    float* __restrict__ Aagg, float* __restrict__ Bagg) {
  int chunk = blockIdx.x;
  int p = blockIdx.y * 256 + threadIdx.x;  // 0..1535
  int c = p >> 4, n = p & 15;
  float An = Aneg[c * 16 + n];
  float h = 0.0f, Ap = 1.0f;
  int l0 = chunk * CLEN;
  for (int i = 0; i < CLEN; ++i) {
    int l = l0 + i;
    float a = __expf(dtv[(size_t)l * 96 + c] * An);
    float b = dtxc[(size_t)l * 96 + c] * dbc2[(size_t)l * 128 + 96 + n];
    h = fmaf(a, h, b);
    Ap *= a;
  }
  Aagg[chunk * 1536 + p] = Ap;
  Bagg[chunk * 1536 + p] = h;
}

// Pass 2: exclusive scan of the 64 chunk aggregates -> carry-in per chunk
__global__ __launch_bounds__(512) void k_scan2(
    const float* __restrict__ Aagg, const float* __restrict__ Bagg,
    float* __restrict__ hinit) {
  for (int p = threadIdx.x; p < 1536; p += 512) {
    float h = 0.0f;
    for (int ch = 0; ch < NCHUNK; ++ch) {
      hinit[ch * 1536 + p] = h;
      h = fmaf(Aagg[ch * 1536 + p], h, Bagg[ch * 1536 + p]);
    }
  }
}

// Pass 3: replay with carry-in, reduce 16 states via lane shuffles, gate, emit f16
__global__ __launch_bounds__(256) void k_scan3(
    const float* __restrict__ dtv, const float* __restrict__ dtxc,
    const float* __restrict__ dbc2, const float* __restrict__ Aneg,
    const float* __restrict__ hinit, const float* __restrict__ xc,
    const float* __restrict__ szg, const float* __restrict__ Dp,
    _Float16* __restrict__ yh) {
  int chunk = blockIdx.x;
  int p = blockIdx.y * 256 + threadIdx.x;
  int c = p >> 4, n = p & 15;
  float An = Aneg[c * 16 + n];
  float h = hinit[chunk * 1536 + p];
  float Dc = Dp[c];
  int l0 = chunk * CLEN;
  for (int i = 0; i < CLEN; ++i) {
    int l = l0 + i;
    float a = __expf(dtv[(size_t)l * 96 + c] * An);
    float b = dtxc[(size_t)l * 96 + c] * dbc2[(size_t)l * 128 + 96 + n];
    h = fmaf(a, h, b);
    float contrib = h * dbc2[(size_t)l * 128 + 112 + n];
    contrib += __shfl_xor(contrib, 1, 16);
    contrib += __shfl_xor(contrib, 2, 16);
    contrib += __shfl_xor(contrib, 4, 16);
    contrib += __shfl_xor(contrib, 8, 16);
    if (n == 0) {
      float yv = (contrib + Dc * xc[(size_t)l * 96 + c]) * szg[(size_t)l * 96 + c];
      yh[(size_t)l * 96 + c] = (_Float16)yv;
    }
  }
}

// ---------------------------------------------------------------------------
// Final scatter: reference does NOT undo the diag permutation on output; the
// sequence maps row-major to (X, Y, Z), then transposed back to (d, Z, Y, X).
__global__ __launch_bounds__(256) void k_scatter(
    const float* __restrict__ yout, float* __restrict__ out) {
  int idx = blockIdx.x * 256 + threadIdx.x;
  if (idx >= L_TOK * DIMC) return;
  int l = idx / DIMC, c = idx - l * DIMC;
  int xv = l >> 10, k = l & 1023;
  int yo = k >> 5, zo = k & 31;
  out[((c * 32 + zo) * 32 + yo) * 16 + xv] = yout[idx];
}

// ---------------------------------------------------------------------------
extern "C" void kernel_launch(void* const* d_in, const int* in_sizes, int n_in,
                              void* d_out, int out_size, void* d_ws, size_t ws_size,
                              hipStream_t stream) {
  (void)in_sizes; (void)n_in; (void)out_size; (void)ws_size;
  const float* x     = (const float*)d_in[0];
  const float* lnw   = (const float*)d_in[1];
  const float* lnb   = (const float*)d_in[2];
  const float* inw   = (const float*)d_in[3];
  const float* convw = (const float*)d_in[4];
  const float* convb = (const float*)d_in[5];
  const float* xpw   = (const float*)d_in[6];
  const float* dtw   = (const float*)d_in[7];
  const float* dtb   = (const float*)d_in[8];
  const float* alog  = (const float*)d_in[9];
  const float* Dp    = (const float*)d_in[10];
  const float* outw  = (const float*)d_in[11];
  float* out = (float*)d_out;

  uint8_t* base = (uint8_t*)d_ws;
  size_t off = 0;
  auto alloc = [&](size_t bytes) -> void* {
    void* p = base + off;
    off = (off + bytes + 255) & ~(size_t)255;
    return p;
  };
  int*      perm = (int*)     alloc(1024 * 4);
  _Float16* W1   = (_Float16*)alloc(192 * 64 * 2);
  _Float16* W2   = (_Float16*)alloc(128 * 96 * 2);
  _Float16* W3   = (_Float16*)alloc(48 * 96 * 2);
  float*    Aneg = (float*)   alloc(1536 * 4);
  _Float16* xnh  = (_Float16*)alloc((size_t)L_TOK * 64 * 2);
  float*    xz   = (float*)   alloc((size_t)L_TOK * 192 * 4);
  float*    xc   = (float*)   alloc((size_t)L_TOK * 96 * 4);
  _Float16* xch  = (_Float16*)alloc((size_t)L_TOK * 96 * 2);
  float*    szg  = (float*)   alloc((size_t)L_TOK * 96 * 4);
  float*    dbc2 = (float*)   alloc((size_t)L_TOK * 128 * 4);
  float*    dtv  = (float*)   alloc((size_t)L_TOK * 96 * 4);
  float*    dtxc = (float*)   alloc((size_t)L_TOK * 96 * 4);
  float*    Aagg = (float*)   alloc((size_t)NCHUNK * 1536 * 4);
  float*    Bagg = (float*)   alloc((size_t)NCHUNK * 1536 * 4);
  float*    hin  = (float*)   alloc((size_t)NCHUNK * 1536 * 4);
  _Float16* yh   = (_Float16*)alloc((size_t)L_TOK * 96 * 2);
  float*    yout = (float*)   alloc((size_t)L_TOK * 48 * 4);

  k_perm<<<1, 1024, 0, stream>>>(perm);
  k_prep<<<1, 256, 0, stream>>>(inw, xpw, dtw, outw, alog, W1, W2, W3, Aneg);
  k_gather_ln<<<64, 256, 0, stream>>>(x, lnw, lnb, perm, xnh);
  // in_proj: (16384 x 64) * (192 x 64)^T -> (16384 x 192)
  gemm_wmma<<<1536, 256, 0, stream>>>(xnh, 64, W1, 64, xz, 192, 1024, 12, 2);
  k_conv<<<6144, 256, 0, stream>>>(xz, convw, convb, xc, xch, szg);
  // fused x_proj+dt_proj: (16384 x 96) * (128 x 96)^T -> (16384 x 128)
  gemm_wmma<<<1024, 256, 0, stream>>>(xch, 96, W2, 96, dbc2, 128, 1024, 8, 3);
  k_dt<<<6144, 256, 0, stream>>>(dbc2, dtb, xc, dtv, dtxc);
  k_scan1<<<dim3(NCHUNK, 6), 256, 0, stream>>>(dtv, dtxc, dbc2, Aneg, Aagg, Bagg);
  k_scan2<<<1, 512, 0, stream>>>(Aagg, Bagg, hin);
  k_scan3<<<dim3(NCHUNK, 6), 256, 0, stream>>>(dtv, dtxc, dbc2, Aneg, hin, xc, szg, Dp, yh);
  // out_proj: (16384 x 96) * (48 x 96)^T -> (16384 x 48)
  gemm_wmma<<<384, 256, 0, stream>>>(yh, 96, W3, 96, yout, 48, 1024, 3, 3);
  k_scatter<<<3072, 256, 0, stream>>>(yout, out);
}
